// HeterSumGraph_71433896067330
// MI455X (gfx1250) — matrix operand order repaired and checked by reading, not compiled, not run on hardware
//
#include <hip/hip_runtime.h>

// ---------------------------------------------------------------------------
// HeterSumGraph forward on MI455X (gfx1250, wave32).
// Pipeline per GAT phase:
//   gemm128_wmma (h = X @ W)            -> v_wmma_f32_16x16x32_f16
//   node_scores  (s_src/s_dst = h @ a)  -> wave32 shuffle reduction
//   init_phase   (temp=bias, emax=-inf, denom=0)
//   edge_logits  (leaky_relu + atomic segment max)
//   edge_soft    (exp + atomic segment sum)
//   edge_scatter (wave-per-edge alpha*h[src] -> atomicAdd temp[dst])
//   gemm128_wmma ((temp + Xres) @ W_lin + b_lin) fused residual+bias
// ---------------------------------------------------------------------------

typedef __attribute__((ext_vector_type(16))) _Float16 v16h;
typedef __attribute__((ext_vector_type(8)))  _Float16 v8h;
typedef __attribute__((ext_vector_type(8)))  float    v8f;

#define DK 128           // feature dim (hard requirement of this kernel set)
#define BM 64            // rows per block in the WMMA GEMM
#define LDP 136          // LDS row pitch in halves (keeps 16B alignment, depads banks)

// ---------------------------------------------------------------------------
// C[M,128] = (A + R?) @ B[128,128] + bias?   (R, bias may be nullptr)
// 256 threads = 8 waves; wave w owns N-tile w (16 cols), loops 4 M-subtiles.
// ---------------------------------------------------------------------------
__global__ __launch_bounds__(256) void gemm128_wmma(
    const float* __restrict__ A, const float* __restrict__ R,
    const float* __restrict__ B, const float* __restrict__ bias,
    float* __restrict__ C, int M)
{
    __shared__ __align__(16) _Float16 Alds[BM][LDP];   // [m][k]
    __shared__ __align__(16) _Float16 Blds[DK][LDP];   // [n][k] (transposed)

    const int t  = threadIdx.x;
    const int m0 = blockIdx.x * BM;
    const bool fullTile = (m0 + BM) <= M;   // uniform: only last block is partial

    // Stage B transposed as f16: B is [k][n] row-major in memory.
    for (int i = t; i < DK * DK; i += 256) {
        int k = i >> 7, n = i & 127;
        Blds[n][k] = (_Float16)B[i];
    }
    // Stage A (+ optional residual) as f16 row-major. Uniform branch selects
    // one of three tight, branch-free staging loops.
    if (fullTile) {
        if (R) {
            for (int i = t; i < BM * DK; i += 256) {
                int r = i >> 7, c = i & 127;
                size_t idx = (size_t)(m0 + r) * DK + c;
                Alds[r][c] = (_Float16)(A[idx] + R[idx]);
            }
        } else {
            for (int i = t; i < BM * DK; i += 256) {
                int r = i >> 7, c = i & 127;
                Alds[r][c] = (_Float16)A[(size_t)(m0 + r) * DK + c];
            }
        }
    } else {
        for (int i = t; i < BM * DK; i += 256) {
            int r = i >> 7, c = i & 127;
            int m = m0 + r; if (m >= M) m = M - 1;
            float v = A[(size_t)m * DK + c];
            if (R) v += R[(size_t)m * DK + c];
            Alds[r][c] = (_Float16)v;
        }
    }
    __syncthreads();

    const int wave  = t >> 5;
    const int lane  = t & 31;
    const int nb    = wave << 4;        // n_base for this wave
    const int lhalf = lane >> 4;        // 0: lanes 0-15, 1: lanes 16-31
    const int l15   = lane & 15;
    const int col   = nb + l15;
    const float badd = bias ? bias[col] : 0.0f;   // hoisted, loop-invariant

    #pragma unroll
    for (int ms = 0; ms < 4; ++ms) {
        const int mb = ms << 4;
        v8f acc = {};
        #pragma unroll
        for (int k0 = 0; k0 < DK; k0 += 32) {
            // A fragment (16x32 f16): lane<16 holds K={k0..k0+7, k0+16..k0+23}
            //                         lane>=16 holds K shifted by +8.
            const int arow = mb + l15;
            const int ka   = k0 + (lhalf << 3);
            v8h alo = *(const v8h*)&Alds[arow][ka];
            v8h ahi = *(const v8h*)&Alds[arow][ka + 16];
            v16h af = __builtin_shufflevector(alo, ahi,
                        0,1,2,3,4,5,6,7,8,9,10,11,12,13,14,15);
            // B fragment (32x16 f16): lane<16 -> col=l15, K={k0..k0+7, k0+16..k0+23};
            //                         lane>=16 -> same col, K shifted by +8.
            const int kb   = k0 + (lhalf << 3);
            v8h blo = *(const v8h*)&Blds[col - nb + nb][kb];   // == Blds[col][kb]
            v8h bhi = *(const v8h*)&Blds[col][kb + 16];
            v16h bf = __builtin_shufflevector(blo, bhi,
                        0,1,2,3,4,5,6,7,8,9,10,11,12,13,14,15);
            acc = __builtin_amdgcn_wmma_f32_16x16x32_f16(
                    false, af, false, bf, (short)0, acc, false, false);
        }
        // C layout: VGPR r -> (M = r + 8*lhalf, N = l15) within the 16x16 tile.
        const int rbase = m0 + mb + (lhalf << 3);
        if (fullTile) {
            #pragma unroll
            for (int r = 0; r < 8; ++r)
                C[(size_t)(rbase + r) * DK + col] = acc[r] + badd;
        } else {
            #pragma unroll
            for (int r = 0; r < 8; ++r) {
                int m = rbase + r;
                if (m < M) C[(size_t)m * DK + col] = acc[r] + badd;
            }
        }
    }
}

// ---------------------------------------------------------------------------
// Per-node attention scores: one wave per node, float4 per lane, shfl reduce.
// ---------------------------------------------------------------------------
__global__ __launch_bounds__(256) void node_scores(
    const float* __restrict__ h, const float* __restrict__ a_src,
    const float* __restrict__ a_dst, float* __restrict__ s_src,
    float* __restrict__ s_dst, int n)
{
    const int gw   = (blockIdx.x * 256 + threadIdx.x) >> 5;
    const int lane = threadIdx.x & 31;
    if (gw >= n) return;
    const float4 hv = ((const float4*)(h + (size_t)gw * DK))[lane];
    const float4 as = ((const float4*)a_src)[lane];
    const float4 ad = ((const float4*)a_dst)[lane];
    float ss = hv.x*as.x + hv.y*as.y + hv.z*as.z + hv.w*as.w;
    float sd = hv.x*ad.x + hv.y*ad.y + hv.z*ad.z + hv.w*ad.w;
    #pragma unroll
    for (int off = 16; off > 0; off >>= 1) {
        ss += __shfl_xor(ss, off, 32);
        sd += __shfl_xor(sd, off, 32);
    }
    if (lane == 0) { s_src[gw] = ss; s_dst[gw] = sd; }
}

// ---------------------------------------------------------------------------
// Per-phase init: temp rows = bias, emax = -inf, denom = 0.
// ---------------------------------------------------------------------------
__global__ __launch_bounds__(256) void init_phase(
    float* __restrict__ temp, float* __restrict__ emax,
    float* __restrict__ denom, const float* __restrict__ bias, int n)
{
    const int i = blockIdx.x * 256 + threadIdx.x;
    if (i < n * DK) temp[i] = bias[i & (DK - 1)];
    if (i < n) {
        emax[i]  = __int_as_float(0xff800000);  // -inf
        denom[i] = 0.0f;
    }
}

// Sign-split float atomic max (init must be -inf). Works with int max for
// non-negative values and uint min for negative values (IEEE ordering).
__device__ __forceinline__ void atomicMaxF(float* addr, float v) {
    if (v >= 0.0f) atomicMax((int*)addr, __float_as_int(v));
    else           atomicMin((unsigned int*)addr, __float_as_uint(v));
}

// ---------------------------------------------------------------------------
// Pass 1: e = leaky_relu(s_src[src] + s_dst[dst]); segment max over dst.
// ---------------------------------------------------------------------------
__global__ __launch_bounds__(256) void edge_logits(
    const int* __restrict__ srcA, const int* __restrict__ dstA,
    const float* __restrict__ s_src, const float* __restrict__ s_dst,
    float* __restrict__ e, float* __restrict__ emax, int ne)
{
    const int i = blockIdx.x * 256 + threadIdx.x;
    if (i >= ne) return;
    float v = s_src[srcA[i]] + s_dst[dstA[i]];
    v = (v > 0.0f) ? v : 0.2f * v;
    e[i] = v;
    atomicMaxF(&emax[dstA[i]], v);
}

// ---------------------------------------------------------------------------
// Pass 2: e <- exp(e - emax[dst]); denom[dst] += e (segment sum).
// ---------------------------------------------------------------------------
__global__ __launch_bounds__(256) void edge_soft(
    const int* __restrict__ dstA, float* __restrict__ e,
    const float* __restrict__ emax, float* __restrict__ denom, int ne)
{
    const int i = blockIdx.x * 256 + threadIdx.x;
    if (i >= ne) return;
    const int d = dstA[i];
    const float ex = __expf(e[i] - emax[d]);
    e[i] = ex;
    atomicAdd(&denom[d], ex);
}

// ---------------------------------------------------------------------------
// Pass 3: temp[dst] += alpha * h[src].  One wave32 per edge: lane owns a
// float4 chunk of the 128-dim row (coalesced 512B gather, 4 atomics/lane).
// ---------------------------------------------------------------------------
__global__ __launch_bounds__(256) void edge_scatter(
    const int* __restrict__ srcA, const int* __restrict__ dstA,
    const float* __restrict__ ex, const float* __restrict__ denom,
    const float* __restrict__ h, float* __restrict__ temp, int ne)
{
    const int gw   = (blockIdx.x * 256 + threadIdx.x) >> 5;
    const int lane = threadIdx.x & 31;
    if (gw >= ne) return;
    const int s = srcA[gw];
    const int d = dstA[gw];
    const float alpha = ex[gw] / (denom[d] + 1e-16f);
    const float4 hv = ((const float4*)(h + (size_t)s * DK))[lane];
    float* o = temp + (size_t)d * DK + lane * 4;
    atomicAdd(o + 0, alpha * hv.x);
    atomicAdd(o + 1, alpha * hv.y);
    atomicAdd(o + 2, alpha * hv.z);
    atomicAdd(o + 3, alpha * hv.w);
}

// ---------------------------------------------------------------------------
// Launch. Inputs (setup_inputs order):
//  0 Xw  1 Xs  2 edge_index(2,NE)  3 W_s2w 4 a_src_s2w 5 a_dst_s2w 6 b_s2w
//  7 W_w2s 8 a_src_w2s 9 a_dst_w2s 10 b_w2s 11 W_lin 12 b_lin
// Output: [hidden_word_next (N*128) | hidden_sent_next (N*128)]
// ---------------------------------------------------------------------------
extern "C" void kernel_launch(void* const* d_in, const int* in_sizes, int n_in,
                              void* d_out, int out_size, void* d_ws, size_t ws_size,
                              hipStream_t stream)
{
    (void)n_in; (void)out_size; (void)ws_size;
    const float* Xw       = (const float*)d_in[0];
    const float* Xs       = (const float*)d_in[1];
    const int*   ei       = (const int*)  d_in[2];
    const float* W_s2w    = (const float*)d_in[3];
    const float* a_src_s2w= (const float*)d_in[4];
    const float* a_dst_s2w= (const float*)d_in[5];
    const float* b_s2w    = (const float*)d_in[6];
    const float* W_w2s    = (const float*)d_in[7];
    const float* a_src_w2s= (const float*)d_in[8];
    const float* a_dst_w2s= (const float*)d_in[9];
    const float* b_w2s    = (const float*)d_in[10];
    const float* W_lin    = (const float*)d_in[11];
    const float* b_lin    = (const float*)d_in[12];

    const int N  = in_sizes[0] / DK;
    const int NE = in_sizes[2] / 2;
    const int* src = ei;        // edge_index[0]
    const int* dst = ei + NE;   // edge_index[1]

    // Workspace layout (floats), reused across the two sequential phases.
    float* ws    = (float*)d_ws;
    float* h     = ws;                              // N*128
    float* temp  = h    + (size_t)N * DK;           // N*128
    float* e     = temp + (size_t)N * DK;           // NE
    float* ssrc  = e    + (size_t)NE;               // N
    float* sdst  = ssrc + N;                        // N
    float* emax  = sdst + N;                        // N
    float* denom = emax + N;                        // N

    float* out_word = (float*)d_out;
    float* out_sent = out_word + (size_t)N * DK;

    const dim3 blk(256);
    const int gGemm  = (N + BM - 1) / BM;
    const int gNodeW = (N + 7) / 8;          // wave-per-node
    const int gInit  = (N * DK + 255) / 256;
    const int gEdge  = (NE + 255) / 256;
    const int gEdgeW = (NE + 7) / 8;         // wave-per-edge

    // -------- Phase A: sent2word (edges src->dst, features Xs) ------------
    gemm128_wmma<<<gGemm, blk, 0, stream>>>(Xs, nullptr, W_s2w, nullptr, h, N);
    node_scores <<<gNodeW, blk, 0, stream>>>(h, a_src_s2w, a_dst_s2w, ssrc, sdst, N);
    init_phase  <<<gInit, blk, 0, stream>>>(temp, emax, denom, b_s2w, N);
    edge_logits <<<gEdge, blk, 0, stream>>>(src, dst, ssrc, sdst, e, emax, NE);
    edge_soft   <<<gEdge, blk, 0, stream>>>(dst, e, emax, denom, NE);
    edge_scatter<<<gEdgeW, blk, 0, stream>>>(src, dst, e, denom, h, temp, NE);
    // hidden_word = (temp_word + Xw) @ W_lin + b_lin
    gemm128_wmma<<<gGemm, blk, 0, stream>>>(temp, Xw, W_lin, b_lin, out_word, N);

    // -------- Phase B: word2sent (edges dst->src, features Xw) ------------
    gemm128_wmma<<<gGemm, blk, 0, stream>>>(Xw, nullptr, W_w2s, nullptr, h, N);
    node_scores <<<gNodeW, blk, 0, stream>>>(h, a_src_w2s, a_dst_w2s, ssrc, sdst, N);
    init_phase  <<<gInit, blk, 0, stream>>>(temp, emax, denom, b_w2s, N);
    edge_logits <<<gEdge, blk, 0, stream>>>(dst, src, ssrc, sdst, e, emax, NE);
    edge_soft   <<<gEdge, blk, 0, stream>>>(src, e, emax, denom, NE);
    edge_scatter<<<gEdgeW, blk, 0, stream>>>(dst, src, e, denom, h, temp, NE);
    // hidden_sent = (temp_sent + Xs) @ W_lin + b_lin
    gemm128_wmma<<<gGemm, blk, 0, stream>>>(temp, Xs, W_lin, b_lin, out_sent, N);
}
